// SConv2dAvg_91087666413981
// MI455X (gfx1250) — matrix operand order repaired and checked by reading, not compiled, not run on hardware
//
#include <hip/hip_runtime.h>

// ---------------------------------------------------------------------------
// SConv2dAvg on MI455X (gfx1250): implicit GEMM, bf16x3 WMMA emulation.
//   M = O = 256, N = B*oh*ow = 32768, K = C*kh*kw = 1152
//   K ordered k = (i*3+j)*128 + c (tap-major, channel-contiguous).
//   WG = 256 threads (8 wave32, 2x4), 128(M) x 128(N) tile, BK = 32.
//   Weights pre-split f32 -> bf16 hi/lo into workspace ([o][tap][c] layout),
//   then staged to LDS with CDNA5 async global->LDS b128 copies (ASYNCcnt).
//   Patches gathered f32, split hi/lo, packed, stored as ds_store_b128.
// ---------------------------------------------------------------------------

typedef __attribute__((ext_vector_type(16))) __bf16        v16bf;
typedef __attribute__((ext_vector_type(8)))  __bf16        v8bf;
typedef __attribute__((ext_vector_type(8)))  float         v8f;
typedef __attribute__((ext_vector_type(4)))  float         v4f;
typedef __attribute__((ext_vector_type(4)))  unsigned int  v4u;

#define C_IN   128
#define O_OUT  256
#define HW_IN  64
#define NPIX   32768            // 32 * 32 * 32
#define LDSK   40               // padded K stride: 80B -> conflict-free b128
#define WSPLIT_ELEMS (O_OUT * 9 * C_IN)   // 294912 per hi/lo plane

__device__ __forceinline__ unsigned short f32_to_bf16_rne(float f) {
    unsigned int u = __float_as_uint(f);
    unsigned int r = u + 0x7FFFu + ((u >> 16) & 1u);
    return (unsigned short)(r >> 16);
}
__device__ __forceinline__ float bf16_hi_to_f32(unsigned short h) {
    return __uint_as_float(((unsigned int)h) << 16);
}

// ---- CDNA5 async global->LDS copy (tracked by ASYNCcnt) -------------------
__device__ __forceinline__ void async_ld16(const void* gsrc, void* ldst) {
    unsigned lds = (unsigned)(uintptr_t)ldst;          // low 32b = LDS offset
    unsigned long long ga = (unsigned long long)(uintptr_t)gsrc;
    asm volatile("global_load_async_to_lds_b128 %0, %1, off"
                 :: "v"(lds), "v"(ga) : "memory");
}
__device__ __forceinline__ void wait_async0() {
    asm volatile("s_wait_asynccnt 0x0" ::: "memory");
}

// Assemble 16-element bf16 fragment from two 8-element runs in LDS.
// A-frag (16x32 MxK): o0 = (lane>=16)?8:0,  gap = 16
// B-frag (32x16 KxN stored [N][K]): o0 = (lane>=16)?8:0, gap = 16
__device__ __forceinline__ v16bf load_frag(const unsigned short* p, int o0, int gap) {
    v8bf a = *(const v8bf*)(p + o0);
    v8bf b = *(const v8bf*)(p + o0 + gap);
    v16bf r;
#pragma unroll
    for (int i = 0; i < 8; ++i) { r[i] = a[i]; r[i + 8] = b[i]; }
    return r;
}

// ---- prep: split weights f32 -> bf16 hi/lo in [o][tap][c] layout ----------
__global__ __launch_bounds__(256) void
weight_split_kernel(const float* __restrict__ weight,
                    unsigned short* __restrict__ wsHi,
                    unsigned short* __restrict__ wsLo)
{
    int idx = blockIdx.x * 256 + threadIdx.x;   // dest index: (o*9 + ij)*128 + c
    int c  = idx & 127;
    int t  = idx >> 7;          // o*9 + ij
    int ij = t % 9;
    int o  = t / 9;
    float w = weight[(o * C_IN + c) * 9 + ij];
    unsigned short hi = f32_to_bf16_rne(w);
    wsHi[idx] = hi;
    wsLo[idx] = f32_to_bf16_rne(w - bf16_hi_to_f32(hi));
}

// ---- main kernel ----------------------------------------------------------
template <bool USE_WS>
__global__ __launch_bounds__(256) void
sconv2davg_wmma_bf16x3(const float* __restrict__ input,   // (32,128,64,64)
                       const float* __restrict__ weight,  // (256,128,3,3)
                       const float* __restrict__ bias,    // (256)
                       const int*   __restrict__ selh,    // (32,32)
                       const int*   __restrict__ selw,    // (32,32)
                       const unsigned short* __restrict__ wsHi,
                       const unsigned short* __restrict__ wsLo,
                       float*       __restrict__ out)     // (32,256,32,32)
{
    __shared__ __align__(16) unsigned short sWhi[128 * LDSK];
    __shared__ __align__(16) unsigned short sWlo[128 * LDSK];
    __shared__ __align__(16) unsigned short sPhi[128 * LDSK];
    __shared__ __align__(16) unsigned short sPlo[128 * LDSK];

    const int tid  = threadIdx.x;
    const int lane = tid & 31;
    const int wave = tid >> 5;
    const int wm   = wave & 1;     // M offset wm*64
    const int wn   = wave >> 1;    // N offset wn*32

    const int tileN = blockIdx.x;  // 0..255
    const int tileM = blockIdx.y;  // 0..1

    // staging assignment: (tile row, 16-element K half)
    const int prow  = tid >> 1;         // 0..127
    const int khalf = (tid & 1) * 16;   // 0 or 16

    // pixel this thread stages
    const int pix = tileN * 128 + prow;
    const int pb  = pix >> 10;
    const int phw = pix & 1023;
    const int ph  = phw >> 5;
    const int pw  = phw & 31;
    const int rh  = selh[phw] + 2 * ph;
    const int rw  = selw[phw] + 2 * pw;
    const float* pbase = input + (long)pb * C_IN * HW_IN * HW_IN;

    const int orow = tileM * 128 + prow;

    v8f acc[4][2];
#pragma unroll
    for (int mi = 0; mi < 4; ++mi)
#pragma unroll
        for (int ni = 0; ni < 2; ++ni)
#pragma unroll
            for (int r = 0; r < 8; ++r) acc[mi][ni][r] = 0.0f;

    const int aoff = (lane >= 16) ? 8  : 0;
    const int boff = (lane >= 16) ? 8  : 0;
    const int arow = wm * 64 + (lane & 15);
    const int brow = wn * 32 + (lane & 15);

    // LDS destinations for this thread's staged 16 elements (32B each)
    unsigned short* dWh = &sWhi[prow * LDSK + khalf];
    unsigned short* dWl = &sWlo[prow * LDSK + khalf];
    v4u* dPh = (v4u*)&sPhi[prow * LDSK + khalf];
    v4u* dPl = (v4u*)&sPlo[prow * LDSK + khalf];

    // ---- K loop: 9 taps x 4 channel blocks of 32 ----
    for (int ij = 0; ij < 9; ++ij) {
        const int ki = ij / 3, kj = ij % 3;
        const int row = rh + ki - 1;
        const int col = rw + kj - 1;
        const bool inb = ((unsigned)row < (unsigned)HW_IN) &&
                         ((unsigned)col < (unsigned)HW_IN);
        const float* src = pbase + row * HW_IN + col;
        const unsigned short* wH =
            USE_WS ? (wsHi + ((long)orow * 9 + ij) * C_IN) : nullptr;
        const unsigned short* wL =
            USE_WS ? (wsLo + ((long)orow * 9 + ij) * C_IN) : nullptr;

        for (int cb = 0; cb < 4; ++cb) {
            const int cbase = cb * 32 + khalf;   // this thread's 16 channels

            __syncthreads();   // prior slice fully consumed before restage

            // ---- weight staging ----
            if (USE_WS) {
                // pre-split bf16, contiguous: async DMA 4 x 16B into LDS
                async_ld16(wH + cbase,      dWh);
                async_ld16(wH + cbase + 8,  dWh + 8);
                async_ld16(wL + cbase,      dWl);
                async_ld16(wL + cbase + 8,  dWl + 8);
            } else {
                const float* wsrc = weight + ((long)orow * C_IN + cbase) * 9 + ij;
                unsigned int hp[8], lp[8];
#pragma unroll
                for (int q = 0; q < 8; ++q) {
                    float w0 = wsrc[(2 * q) * 9];
                    float w1 = wsrc[(2 * q + 1) * 9];
                    unsigned short h0 = f32_to_bf16_rne(w0);
                    unsigned short h1 = f32_to_bf16_rne(w1);
                    unsigned short l0 = f32_to_bf16_rne(w0 - bf16_hi_to_f32(h0));
                    unsigned short l1 = f32_to_bf16_rne(w1 - bf16_hi_to_f32(h1));
                    hp[q] = (unsigned)h0 | ((unsigned)h1 << 16);
                    lp[q] = (unsigned)l0 | ((unsigned)l1 << 16);
                }
                v4u* dh = (v4u*)dWh; v4u* dl = (v4u*)dWl;
                dh[0] = (v4u){hp[0], hp[1], hp[2], hp[3]};
                dh[1] = (v4u){hp[4], hp[5], hp[6], hp[7]};
                dl[0] = (v4u){lp[0], lp[1], lp[2], lp[3]};
                dl[1] = (v4u){lp[4], lp[5], lp[6], lp[7]};
            }

            // ---- patch staging: gather 16 strided f32, split, pack, b128 ----
            {
                unsigned int hp[8], lp[8];
#pragma unroll
                for (int q = 0; q < 8; ++q) {
                    float x0 = inb ? src[(long)(cbase + 2 * q)     * (HW_IN * HW_IN)] : 0.0f;
                    float x1 = inb ? src[(long)(cbase + 2 * q + 1) * (HW_IN * HW_IN)] : 0.0f;
                    unsigned short h0 = f32_to_bf16_rne(x0);
                    unsigned short h1 = f32_to_bf16_rne(x1);
                    unsigned short l0 = f32_to_bf16_rne(x0 - bf16_hi_to_f32(h0));
                    unsigned short l1 = f32_to_bf16_rne(x1 - bf16_hi_to_f32(h1));
                    hp[q] = (unsigned)h0 | ((unsigned)h1 << 16);
                    lp[q] = (unsigned)l0 | ((unsigned)l1 << 16);
                }
                dPh[0] = (v4u){hp[0], hp[1], hp[2], hp[3]};
                dPh[1] = (v4u){hp[4], hp[5], hp[6], hp[7]};
                dPl[0] = (v4u){lp[0], lp[1], lp[2], lp[3]};
                dPl[1] = (v4u){lp[4], lp[5], lp[6], lp[7]};
            }

            if (USE_WS) wait_async0();   // ASYNCcnt -> 0 before barrier
            __syncthreads();

            // ---- compute: 4x2 fragments, 3 WMMAs each (bf16x3) ----
            v16bf bhi[2], blo[2];
#pragma unroll
            for (int ni = 0; ni < 2; ++ni) {
                bhi[ni] = load_frag(&sPhi[(brow + ni * 16) * LDSK], boff, 16);
                blo[ni] = load_frag(&sPlo[(brow + ni * 16) * LDSK], boff, 16);
            }
#pragma unroll
            for (int mi = 0; mi < 4; ++mi) {
                v16bf ahi = load_frag(&sWhi[(arow + mi * 16) * LDSK], aoff, 16);
                v16bf alo = load_frag(&sWlo[(arow + mi * 16) * LDSK], aoff, 16);
#pragma unroll
                for (int ni = 0; ni < 2; ++ni) {
                    acc[mi][ni] = __builtin_amdgcn_wmma_f32_16x16x32_bf16(
                        false, ahi, false, bhi[ni], (short)0, acc[mi][ni], false, false);
                    acc[mi][ni] = __builtin_amdgcn_wmma_f32_16x16x32_bf16(
                        false, ahi, false, blo[ni], (short)0, acc[mi][ni], false, false);
                    acc[mi][ni] = __builtin_amdgcn_wmma_f32_16x16x32_bf16(
                        false, alo, false, bhi[ni], (short)0, acc[mi][ni], false, false);
                }
            }
        }
    }

    // ---- epilogue: C/D layout -> (B,O,oh,ow), add bias --------------------
    const int lhalf = (lane >= 16) ? 8 : 0;   // VGPR r -> M = r + lhalf
    const int ncol  = lane & 15;              // lane -> N column
    const int bidx  = (tileN * 128) >> 10;    // constant batch per block
    const int hwb   = (tileN * 128 & 1023) + wn * 32 + ncol;
    float* obase = out + (long)bidx * O_OUT * 1024;

#pragma unroll
    for (int mi = 0; mi < 4; ++mi) {
        const int om = tileM * 128 + wm * 64 + mi * 16 + lhalf;
        v4f b0 = *(const v4f*)&bias[om];
        v4f b1 = *(const v4f*)&bias[om + 4];
#pragma unroll
        for (int ni = 0; ni < 2; ++ni) {
            const int hw = hwb + ni * 16;
#pragma unroll
            for (int r = 0; r < 8; ++r) {
                float bv = (r < 4) ? b0[r & 3] : b1[r & 3];
                obase[(long)(om + r) * 1024 + hw] = acc[mi][ni][r] + bv;
            }
        }
    }
}

extern "C" void kernel_launch(void* const* d_in, const int* in_sizes, int n_in,
                              void* d_out, int out_size, void* d_ws, size_t ws_size,
                              hipStream_t stream) {
    const float* input  = (const float*)d_in[0];
    const float* weight = (const float*)d_in[1];
    const float* bias   = (const float*)d_in[2];
    const int*   selh   = (const int*)d_in[3];
    const int*   selw   = (const int*)d_in[4];
    float*       out    = (float*)d_out;

    dim3 grid(NPIX / 128, O_OUT / 128);   // (256, 2)
    dim3 block(256);

    const size_t ws_needed = (size_t)WSPLIT_ELEMS * 2 * sizeof(unsigned short);
    if (ws_size >= ws_needed) {
        unsigned short* wsHi = (unsigned short*)d_ws;
        unsigned short* wsLo = wsHi + WSPLIT_ELEMS;
        weight_split_kernel<<<WSPLIT_ELEMS / 256, 256, 0, stream>>>(weight, wsHi, wsLo);
        sconv2davg_wmma_bf16x3<true><<<grid, block, 0, stream>>>(
            input, weight, bias, selh, selw, wsHi, wsLo, out);
    } else {
        sconv2davg_wmma_bf16x3<false><<<grid, block, 0, stream>>>(
            input, weight, bias, selh, selw, nullptr, nullptr, out);
    }
}
